// EncoderBlock_27960237097222
// MI455X (gfx1250) — compile-verified, matrix operand order checked by reading, not run on hardware
//
#include <hip/hip_runtime.h>

// ---------------------------------------------------------------------------
// CDNA5 (gfx1250) transformer encoder block.
// bf16 WMMA for all GEMMs; GEMM operands staged in LDS via gfx1250
// global_load_async_to_lds_b128 (ASYNCcnt), double buffered.
// ---------------------------------------------------------------------------

typedef __bf16 bf16_t;
typedef __attribute__((ext_vector_type(16))) __bf16 bf16x16;
typedef __attribute__((ext_vector_type(8)))  __bf16 bf16x8;
typedef __attribute__((ext_vector_type(8)))  float   f32x8;

#define BB     2
#define SS     2048
#define DMODEL 1024
#define HH     16
#define DKK    64
#define DFF    4096
#define MROWS  (BB * SS)

#define NEG_INF (-3.0e38f)

// LDS tile row stride in bf16 elements: 40 elems = 80 bytes = 20 banks
// (16 consecutive rows -> 16 distinct bank groups for ds_load_b128).
#define TSTRIDE 40

// ---- WMMA wrapper: D = A(16x32 bf16) * B(32x16 bf16) + C(16x16 f32) --------
__device__ __forceinline__ f32x8 wmma_bf16(bf16x16 a, bf16x16 b, f32x8 c) {
  return __builtin_amdgcn_wmma_f32_16x16x32_bf16(
      /*neg_a=*/false, a, /*neg_b=*/false, b,
      /*c_mod=*/(short)0, c, /*reuse_a=*/false, /*reuse_b=*/false);
}

// ---- fragment loader --------------------------------------------------------
// ISA 7.12.2, 16-bit A 16x32: lane L holds row M=L%16; lanes 0-15 hold K
// chunks {0..7, 16..23}, lanes 16-31 hold {8..15, 24..31}. Caller passes
// p = row_base + k0 + (lane>>4)*8; we fetch p[0..7] and p[16..23] (2x b128).
// B operand (32x16, column n has K contiguous) uses the identical pattern.
__device__ __forceinline__ bf16x16 load_frag(const bf16_t* p) {
  bf16x8 lo = *(const bf16x8*)p;
  bf16x8 hi = *(const bf16x8*)(p + 16);
  bf16x16 r;
#pragma unroll
  for (int i = 0; i < 8; ++i) { r[i] = lo[i]; r[i + 8] = hi[i]; }
  return r;
}

// ---- gfx1250 async copy: 16B per lane, global -> LDS, tracked by ASYNCcnt --
// Generic pointers to __shared__ carry the LDS byte offset in their low 32
// bits, which is what VDST must supply.
__device__ __forceinline__ void async_b128(const bf16_t* gsrc, const bf16_t* ldst) {
  unsigned lds_off = (unsigned)(unsigned long long)(uintptr_t)ldst;
  asm volatile("global_load_async_to_lds_b128 %0, %1, off"
               :: "v"(lds_off), "v"(gsrc)
               : "memory");
}

// ---- intra-half (16-lane) reductions for softmax ---------------------------
__device__ __forceinline__ float half_max(float v) {
#pragma unroll
  for (int m = 1; m < 16; m <<= 1) v = fmaxf(v, __shfl_xor(v, m, 32));
  return v;
}
__device__ __forceinline__ float half_sum(float v) {
#pragma unroll
  for (int m = 1; m < 16; m <<= 1) v += __shfl_xor(v, m, 32);
  return v;
}

// ---------------------------------------------------------------------------
// fp32 -> bf16 conversion (weights)
// ---------------------------------------------------------------------------
__global__ __launch_bounds__(256) void cvt_bf16_kernel(
    const float* __restrict__ src, bf16_t* __restrict__ dst, int n) {
  int i = blockIdx.x * 256 + threadIdx.x;
  if (i < n) dst[i] = (bf16_t)src[i];
}

// ---------------------------------------------------------------------------
// LayerNorm (fp32 in, bf16 out). One block per row of 1024.
// ---------------------------------------------------------------------------
__global__ __launch_bounds__(256) void ln_bf16_kernel(
    const float* __restrict__ x, const float* __restrict__ g,
    const float* __restrict__ be, bf16_t* __restrict__ y) {
  const int row = blockIdx.x, tid = threadIdx.x;
  const int wid = tid >> 5, lane = tid & 31;
  const float4 v = ((const float4*)(x + (size_t)row * DMODEL))[tid];
  float s  = v.x + v.y + v.z + v.w;
  float s2 = v.x * v.x + v.y * v.y + v.z * v.z + v.w * v.w;
#pragma unroll
  for (int m = 1; m < 32; m <<= 1) {
    s  += __shfl_xor(s,  m, 32);
    s2 += __shfl_xor(s2, m, 32);
  }
  __shared__ float sh1[8], sh2[8];
  if (lane == 0) { sh1[wid] = s; sh2[wid] = s2; }
  __syncthreads();
  float ts = 0.f, ts2 = 0.f;
#pragma unroll
  for (int i = 0; i < 8; ++i) { ts += sh1[i]; ts2 += sh2[i]; }
  const float mu   = ts  * (1.0f / DMODEL);
  const float var  = ts2 * (1.0f / DMODEL) - mu * mu;
  const float rstd = rsqrtf(var + 1e-5f);
  const float4 gv = ((const float4*)g)[tid];
  const float4 bv = ((const float4*)be)[tid];
  bf16_t* yp = y + (size_t)row * DMODEL + tid * 4;
  yp[0] = (bf16_t)((v.x - mu) * rstd * gv.x + bv.x);
  yp[1] = (bf16_t)((v.y - mu) * rstd * gv.y + bv.y);
  yp[2] = (bf16_t)((v.z - mu) * rstd * gv.z + bv.z);
  yp[3] = (bf16_t)((v.w - mu) * rstd * gv.w + bv.w);
}

// ---------------------------------------------------------------------------
// GEMM: C[M,N] = A[M,K](bf16) @ W[N,K](bf16)^T, fused epilogues.
// Block 256 = 8 waves; block tile 128x256; each wave a 64x64 tile (4x4 WMMA).
// K step 32. A/B tiles staged in LDS via async load-to-LDS, double buffered:
//   iter: [issue tile k+1 -> s_wait_asynccnt 6 -> barrier -> compute -> barrier]
// 6 async instructions/wave/tile; previous batch completes in order.
// ---------------------------------------------------------------------------
constexpr int EPI_BF16          = 0;  // write bf16 C
constexpr int EPI_VT            = 1;  // write bf16 C transposed per-head (V)
constexpr int EPI_RESF32        = 2;  // C + resid -> f32
constexpr int EPI_BIAS_RELU_BF16= 3;  // relu(C + bias) -> bf16
constexpr int EPI_BIAS_RES_F32  = 4;  // C + bias + resid -> f32

template <int EPI>
__global__ __launch_bounds__(256) void gemm_bf16_kernel(
    const bf16_t* __restrict__ A, const bf16_t* __restrict__ W,
    int M, int N, int K,
    const float* __restrict__ bias, const float* __restrict__ resid,
    float* __restrict__ outf, bf16_t* __restrict__ outbf) {
  // A tile: 128 rows x 32 K (80B padded rows). B tile: 256 rows x 32 K.
  __shared__ __align__(16) bf16_t abuf[2][128 * TSTRIDE];  // 2 x 10 KB
  __shared__ __align__(16) bf16_t bbuf[2][256 * TSTRIDE];  // 2 x 20 KB

  const int tid  = threadIdx.x;
  const int wid  = tid >> 5;
  const int lane = tid & 31;
  const int lrow = lane & 15;
  const int hi   = lane >> 4;
  const int brow0 = blockIdx.y * 128;
  const int bcol0 = blockIdx.x * 256;
  const int wr0 = (wid >> 2) * 64;   // wave row offset in block tile
  const int wc0 = (wid & 3) * 64;    // wave col offset in block tile

  // cooperative async copy of one K-slab (6 b128 transfers per thread)
  auto issue_tile = [&](int bufi, int kk) {
#pragma unroll
    for (int i = 0; i < 2; ++i) {          // A: 128 rows x 4 chunks = 512
      const int c = tid + 256 * i;
      const int row = c >> 2, seg = c & 3;
      async_b128(A + (size_t)(brow0 + row) * K + kk + seg * 8,
                 &abuf[bufi][row * TSTRIDE + seg * 8]);
    }
#pragma unroll
    for (int i = 0; i < 4; ++i) {          // B: 256 rows x 4 chunks = 1024
      const int c = tid + 256 * i;
      const int row = c >> 2, seg = c & 3;
      async_b128(W + (size_t)(bcol0 + row) * K + kk + seg * 8,
                 &bbuf[bufi][row * TSTRIDE + seg * 8]);
    }
  };

  const f32x8 vzero = {0.f, 0.f, 0.f, 0.f, 0.f, 0.f, 0.f, 0.f};
  f32x8 acc[4][4];
#pragma unroll
  for (int i = 0; i < 4; ++i)
#pragma unroll
    for (int j = 0; j < 4; ++j) acc[i][j] = vzero;

  issue_tile(0, 0);
  int buf = 0;
  for (int kk = 0; kk < K; kk += 32) {
    if (kk + 32 < K) {
      issue_tile(buf ^ 1, kk + 32);
      asm volatile("s_wait_asynccnt 0x6" ::: "memory");  // prev batch landed
    } else {
      asm volatile("s_wait_asynccnt 0x0" ::: "memory");  // last batch landed
    }
    __syncthreads();

    const bf16_t* al = &abuf[buf][(wr0 + lrow) * TSTRIDE + hi * 8];
    const bf16_t* bl = &bbuf[buf][(wc0 + lrow) * TSTRIDE + hi * 8];
    bf16x16 af[4], wf[4];
#pragma unroll
    for (int i = 0; i < 4; ++i) af[i] = load_frag(al + 16 * i * TSTRIDE);
#pragma unroll
    for (int j = 0; j < 4; ++j) wf[j] = load_frag(bl + 16 * j * TSTRIDE);
#pragma unroll
    for (int i = 0; i < 4; ++i)
#pragma unroll
      for (int j = 0; j < 4; ++j) acc[i][j] = wmma_bf16(af[i], wf[j], acc[i][j]);

    __syncthreads();   // release buffer refilled next iteration
    buf ^= 1;
  }

  // Epilogue. C layout: element (row = r + 8*hi, col = lane%16) in VGPR r.
#pragma unroll
  for (int i = 0; i < 4; ++i) {
#pragma unroll
    for (int j = 0; j < 4; ++j) {
      const int col = bcol0 + wc0 + 16 * j + lrow;
#pragma unroll
      for (int r = 0; r < 8; ++r) {
        const int row = brow0 + wr0 + 16 * i + r + 8 * hi;
        float v = acc[i][j][r];
        if constexpr (EPI == EPI_BIAS_RELU_BF16 || EPI == EPI_BIAS_RES_F32)
          v += bias[col];
        if constexpr (EPI == EPI_BIAS_RELU_BF16)
          v = fmaxf(v, 0.0f);
        if constexpr (EPI == EPI_RESF32 || EPI == EPI_BIAS_RES_F32)
          v += resid[(size_t)row * N + col];
        if constexpr (EPI == EPI_BF16 || EPI == EPI_BIAS_RELU_BF16)
          outbf[(size_t)row * N + col] = (bf16_t)v;
        if constexpr (EPI == EPI_VT) {
          // V output transposed per head: [b][h][d][s], s contiguous
          const int bidx = row >> 11, srow = row & 2047;
          const int hh = col >> 6, dd = col & 63;
          outbf[(((size_t)bidx * HH + hh) * DKK + dd) * SS + srow] = (bf16_t)v;
        }
        if constexpr (EPI == EPI_RESF32 || EPI == EPI_BIAS_RES_F32)
          outf[(size_t)row * N + col] = v;
      }
    }
  }
}

// ---------------------------------------------------------------------------
// Flash attention: grid (S/64, B*H), 128 threads = 4 waves.
// Each wave: 16 q rows x d_k=64 output, online softmax over S in 64-col steps.
// Q,K: [b*S + s][h*64 + d] bf16 (row-major).  V: transposed [bh][d][s] bf16.
// ---------------------------------------------------------------------------
__global__ __launch_bounds__(128) void attn_kernel(
    const bf16_t* __restrict__ Q, const bf16_t* __restrict__ Kmat,
    const bf16_t* __restrict__ Vt, const int* __restrict__ mask,
    bf16_t* __restrict__ Ctx) {
  __shared__ __align__(16) bf16_t psh[4][16][80];  // padded stride: bank spread

  const int tid  = threadIdx.x;
  const int wid  = tid >> 5;
  const int lane = tid & 31;
  const int lrow = lane & 15;
  const int hi   = lane >> 4;
  const int bh = blockIdx.y;           // 0..31
  const int b  = bh >> 4;
  const int h  = bh & 15;
  const int q0 = blockIdx.x * 64 + wid * 16;  // this wave's q-row base
  const float scale = 0.125f;          // 1/sqrt(64)

  const f32x8 vzero = {0.f, 0.f, 0.f, 0.f, 0.f, 0.f, 0.f, 0.f};

  // Q A-fragments (resident for whole loop): d chunks [0..31], [32..63]
  const bf16_t* qp = Q + (size_t)(b * SS + q0 + lrow) * DMODEL + h * DKK + hi * 8;
  const bf16x16 qa0 = load_frag(qp);
  const bf16x16 qa1 = load_frag(qp + 32);

  f32x8 O[4];
  float m_i[8], l_i[8];
#pragma unroll
  for (int t = 0; t < 4; ++t) O[t] = vzero;
#pragma unroll
  for (int r = 0; r < 8; ++r) { m_i[r] = NEG_INF; l_i[r] = 0.f; }

  for (int kb = 0; kb < SS; kb += 64) {
    // --- scores: 4 tiles of 16x16, each = Q(16x64) . K_tile(64x16) ----------
    f32x8 st[4];
#pragma unroll
    for (int t = 0; t < 4; ++t) {
      const bf16_t* kp =
          Kmat + (size_t)(b * SS + kb + 16 * t + lrow) * DMODEL + h * DKK + hi * 8;
      bf16x16 kf0 = load_frag(kp);
      bf16x16 kf1 = load_frag(kp + 32);
      f32x8 z = vzero;
      z = wmma_bf16(qa0, kf0, z);
      z = wmma_bf16(qa1, kf1, z);
      st[t] = z;
    }
    // scale + mask
#pragma unroll
    for (int t = 0; t < 4; ++t) {
      const int mv = mask[b * SS + kb + 16 * t + lrow];
#pragma unroll
      for (int r = 0; r < 8; ++r) {
        float v = st[t][r] * scale;
        st[t][r] = (mv == 0) ? NEG_INF : v;
      }
    }
    // --- online softmax update ---------------------------------------------
    float mnew[8], corr[8];
#pragma unroll
    for (int r = 0; r < 8; ++r) {
      float v = fmaxf(fmaxf(st[0][r], st[1][r]), fmaxf(st[2][r], st[3][r]));
      v = half_max(v);
      v = fmaxf(v, m_i[r]);
      mnew[r] = v;
      corr[r] = __expf(m_i[r] - v);
      m_i[r]  = v;
    }
    float rs[8];
#pragma unroll
    for (int r = 0; r < 8; ++r) rs[r] = 0.f;
#pragma unroll
    for (int t = 0; t < 4; ++t)
#pragma unroll
      for (int r = 0; r < 8; ++r) {
        float p = __expf(st[t][r] - mnew[r]);
        st[t][r] = p;
        rs[r] += p;
      }
#pragma unroll
    for (int r = 0; r < 8; ++r) {
      l_i[r] = l_i[r] * corr[r] + half_sum(rs[r]);
    }
#pragma unroll
    for (int t = 0; t < 4; ++t)
#pragma unroll
      for (int r = 0; r < 8; ++r) O[t][r] *= corr[r];

    // --- P(16x64) through LDS: C-layout -> A-layout -------------------------
#pragma unroll
    for (int t = 0; t < 4; ++t)
#pragma unroll
      for (int r = 0; r < 8; ++r)
        psh[wid][r + 8 * hi][16 * t + lrow] = (bf16_t)st[t][r];
    __syncthreads();
    const bf16x16 pa0 = load_frag(&psh[wid][lrow][hi * 8]);
    const bf16x16 pa1 = load_frag(&psh[wid][lrow][32 + hi * 8]);

    // --- O += P . V : V^T rows give contiguous-K B operands -----------------
#pragma unroll
    for (int t2 = 0; t2 < 4; ++t2) {
      const bf16_t* vp =
          Vt + (size_t)(bh * DKK + 16 * t2 + lrow) * SS + kb + hi * 8;
      bf16x16 vf0 = load_frag(vp);
      bf16x16 vf1 = load_frag(vp + 32);
      O[t2] = wmma_bf16(pa0, vf0, O[t2]);
      O[t2] = wmma_bf16(pa1, vf1, O[t2]);
    }
    __syncthreads();
  }

  // --- normalize and write ctx (row-major [m][h*64+d], bf16) ----------------
  float linv[8];
#pragma unroll
  for (int r = 0; r < 8; ++r) linv[r] = 1.0f / l_i[r];
#pragma unroll
  for (int t2 = 0; t2 < 4; ++t2)
#pragma unroll
    for (int r = 0; r < 8; ++r) {
      const size_t row = (size_t)(b * SS + q0 + r + 8 * hi);
      Ctx[row * DMODEL + h * DKK + 16 * t2 + lrow] = (bf16_t)(O[t2][r] * linv[r]);
    }
}

// ---------------------------------------------------------------------------
// Launch
// ---------------------------------------------------------------------------
extern "C" void kernel_launch(void* const* d_in, const int* in_sizes, int n_in,
                              void* d_out, int out_size, void* d_ws, size_t ws_size,
                              hipStream_t stream) {
  const float* x    = (const float*)d_in[0];
  const int*   mask = (const int*)  d_in[1];
  const float* w_q  = (const float*)d_in[2];
  const float* w_k  = (const float*)d_in[3];
  const float* w_v  = (const float*)d_in[4];
  const float* w_o  = (const float*)d_in[5];
  const float* w1   = (const float*)d_in[6];
  const float* b1   = (const float*)d_in[7];
  const float* w2   = (const float*)d_in[8];
  const float* b2   = (const float*)d_in[9];
  const float* g1   = (const float*)d_in[10];
  const float* be1  = (const float*)d_in[11];
  const float* g2   = (const float*)d_in[12];
  const float* be2  = (const float*)d_in[13];

  char* ws = (char*)d_ws;
  size_t off = 0;
  auto take = [&](size_t bytes) -> char* {
    char* p = ws + off;
    off += (bytes + 255) & ~(size_t)255;
    return p;
  };
  bf16_t* wqb = (bf16_t*)take((size_t)DMODEL * DMODEL * 2);
  bf16_t* wkb = (bf16_t*)take((size_t)DMODEL * DMODEL * 2);
  bf16_t* wvb = (bf16_t*)take((size_t)DMODEL * DMODEL * 2);
  bf16_t* wob = (bf16_t*)take((size_t)DMODEL * DMODEL * 2);
  bf16_t* w1b = (bf16_t*)take((size_t)DFF * DMODEL * 2);
  bf16_t* w2b = (bf16_t*)take((size_t)DMODEL * DFF * 2);
  bf16_t* lnb = (bf16_t*)take((size_t)MROWS * DMODEL * 2);
  bf16_t* qb  = (bf16_t*)take((size_t)MROWS * DMODEL * 2);
  bf16_t* kb_ = (bf16_t*)take((size_t)MROWS * DMODEL * 2);
  bf16_t* vtb = (bf16_t*)take((size_t)MROWS * DMODEL * 2);
  bf16_t* ctxb= (bf16_t*)take((size_t)MROWS * DMODEL * 2);
  float*  x1f = (float*) take((size_t)MROWS * DMODEL * 4);
  bf16_t* hb  = (bf16_t*)take((size_t)MROWS * DFF * 2);
  (void)ws_size; (void)in_sizes; (void)n_in; (void)out_size;

  // weights -> bf16
  {
    const int nw = DMODEL * DMODEL;
    cvt_bf16_kernel<<<(nw + 255) / 256, 256, 0, stream>>>(w_q, wqb, nw);
    cvt_bf16_kernel<<<(nw + 255) / 256, 256, 0, stream>>>(w_k, wkb, nw);
    cvt_bf16_kernel<<<(nw + 255) / 256, 256, 0, stream>>>(w_v, wvb, nw);
    cvt_bf16_kernel<<<(nw + 255) / 256, 256, 0, stream>>>(w_o, wob, nw);
    const int nf = DFF * DMODEL;
    cvt_bf16_kernel<<<(nf + 255) / 256, 256, 0, stream>>>(w1, w1b, nf);
    cvt_bf16_kernel<<<(nf + 255) / 256, 256, 0, stream>>>(w2, w2b, nf);
  }

  // LN1 -> bf16
  ln_bf16_kernel<<<MROWS, 256, 0, stream>>>(x, g1, be1, lnb);

  // Q, K (row-major bf16), V (transposed per-head bf16)
  dim3 gemm_grid_d(DMODEL / 256, MROWS / 128);
  gemm_bf16_kernel<EPI_BF16><<<gemm_grid_d, 256, 0, stream>>>(
      lnb, wqb, MROWS, DMODEL, DMODEL, nullptr, nullptr, nullptr, qb);
  gemm_bf16_kernel<EPI_BF16><<<gemm_grid_d, 256, 0, stream>>>(
      lnb, wkb, MROWS, DMODEL, DMODEL, nullptr, nullptr, nullptr, kb_);
  gemm_bf16_kernel<EPI_VT><<<gemm_grid_d, 256, 0, stream>>>(
      lnb, wvb, MROWS, DMODEL, DMODEL, nullptr, nullptr, nullptr, vtb);

  // flash attention -> ctx bf16
  attn_kernel<<<dim3(SS / 64, BB * HH), 128, 0, stream>>>(qb, kb_, vtb, mask, ctxb);

  // out-proj + residual -> x1 (fp32)
  gemm_bf16_kernel<EPI_RESF32><<<gemm_grid_d, 256, 0, stream>>>(
      ctxb, wob, MROWS, DMODEL, DMODEL, nullptr, x, x1f, nullptr);

  // LN2 -> bf16 (reuse lnb)
  ln_bf16_kernel<<<MROWS, 256, 0, stream>>>(x1f, g2, be2, lnb);

  // FFN1: relu(ln2 @ w1^T + b1) -> bf16
  gemm_bf16_kernel<EPI_BIAS_RELU_BF16><<<dim3(DFF / 256, MROWS / 128), 256, 0, stream>>>(
      lnb, w1b, MROWS, DFF, DMODEL, b1, nullptr, nullptr, hb);

  // FFN2: h @ w2^T + b2 + x1 -> d_out (fp32)
  gemm_bf16_kernel<EPI_BIAS_RES_F32><<<gemm_grid_d, 256, 0, stream>>>(
      hb, w2b, MROWS, DMODEL, DFF, b2, x1f, (float*)d_out, nullptr);
}
